// INT8Linear_59468117180901
// MI455X (gfx1250) — 8-bit matrix operand order repaired and checked by reading, compiled, not run on hardware
//
#include <hip/hip_runtime.h>
#include <hip/hip_bf16.h>

typedef __attribute__((ext_vector_type(2))) int v2i;
typedef __attribute__((ext_vector_type(4))) int v4i;
typedef __attribute__((ext_vector_type(8))) int v8i;

#define QMAX 127.0f
#define EPS 1e-8f

__global__ __launch_bounds__(256) void quantize_rows_kernel(
    const float* __restrict__ src, signed char* __restrict__ dst,
    float* __restrict__ scale, int ncols)
{
    const int row  = blockIdx.x;
    const int lane = threadIdx.x & 31;
    const int wave = threadIdx.x >> 5;

    const float4* srow = (const float4*)(src + (size_t)row * ncols);
    float4 v = srow[threadIdx.x];

    float m = fmaxf(fmaxf(fabsf(v.x), fabsf(v.y)), fmaxf(fabsf(v.z), fabsf(v.w)));

#pragma unroll
    for (int off = 16; off > 0; off >>= 1)
        m = fmaxf(m, __shfl_xor(m, off, 32));

    __shared__ float smax[8];
    if (lane == 0) smax[wave] = m;
    __syncthreads();

    float am = smax[0];
#pragma unroll
    for (int i = 1; i < 8; ++i) am = fmaxf(am, smax[i]);
    am = fmaxf(am, EPS);

    const float sc  = am / QMAX;
    const float inv = QMAX / am;
    if (threadIdx.x == 0) scale[row] = sc;

    int q0 = (int)rintf(v.x * inv); q0 = min(max(q0, -128), 127);
    int q1 = (int)rintf(v.y * inv); q1 = min(max(q1, -128), 127);
    int q2 = (int)rintf(v.z * inv); q2 = min(max(q2, -128), 127);
    int q3 = (int)rintf(v.w * inv); q3 = min(max(q3, -128), 127);
    unsigned int packed = (unsigned int)(q0 & 0xff) | ((unsigned int)(q1 & 0xff) << 8) |
                          ((unsigned int)(q2 & 0xff) << 16) | ((unsigned int)(q3 & 0xff) << 24);
    ((unsigned int*)(dst + (size_t)row * ncols))[threadIdx.x] = packed;
}

__global__ __launch_bounds__(128) void int8_gemm_wmma_kernel(
    const signed char* __restrict__ xq, const signed char* __restrict__ wq,
    const float* __restrict__ xs, const float* __restrict__ ws,
    const float* __restrict__ bias, float* __restrict__ out,
    int M, int N, int K)
{
    const int lane = threadIdx.x & 31;
    const int wave = threadIdx.x >> 5;
    const int wm   = wave >> 1;
    const int wn   = wave & 1;
    const int half = lane >> 4;
    const int l15  = lane & 15;

    const int blockM = blockIdx.y * 128;
    const int blockN = blockIdx.x * 128;

    const size_t Ksz = (size_t)K;
    const signed char* aBase = xq + (size_t)(blockM + wm * 64 + l15) * Ksz + half * 8;
    const signed char* bBase = wq + (size_t)(blockN + wn * 64 + l15) * Ksz + half * 8;

    v8i acc[4][4] = {};

    for (int k0 = 0; k0 < K; k0 += 64) {
        v8i a[4], b[4];
#pragma unroll
        for (int mi = 0; mi < 4; ++mi) {
            const signed char* p = aBase + (size_t)(mi * 16) * Ksz + k0;
            v4i lo, hi;
            lo.lo = *(const v2i*)(p +  0);
            lo.hi = *(const v2i*)(p + 16);
            hi.lo = *(const v2i*)(p + 32);
            hi.hi = *(const v2i*)(p + 48);
            a[mi].lo = lo; a[mi].hi = hi;
        }
#pragma unroll
        for (int ni = 0; ni < 4; ++ni) {
            const signed char* p = bBase + (size_t)(ni * 16) * Ksz + k0;
            v8i t;
            t.lo.lo = *(const v2i*)(p +  0);
            t.lo.hi = *(const v2i*)(p + 16);
            t.hi.lo = *(const v2i*)(p + 32);
            t.hi.hi = *(const v2i*)(p + 48);
            b[ni] = t;
        }
        if (k0 + 64 < K) {
            __builtin_prefetch(aBase + k0 + 64, 0, 0);
            __builtin_prefetch(bBase + k0 + 64, 0, 0);
        }
#pragma unroll
        for (int mi = 0; mi < 4; ++mi)
#pragma unroll
            for (int ni = 0; ni < 4; ++ni)
                acc[mi][ni] = __builtin_amdgcn_wmma_i32_16x16x64_iu8(
                     true, a[mi],  true, b[ni],
                    acc[mi][ni],  false,  false);
    }

#pragma unroll
    for (int mi = 0; mi < 4; ++mi) {
        const int rbase = blockM + wm * 64 + mi * 16 + half * 8;
        float xsv[8];
#pragma unroll
        for (int e = 0; e < 8; ++e) xsv[e] = xs[rbase + e];
#pragma unroll
        for (int ni = 0; ni < 4; ++ni) {
            const int col = blockN + wn * 64 + ni * 16 + l15;
            const float wsv = ws[col];
            const float bv  = bias[col];
#pragma unroll
            for (int e = 0; e < 8; ++e)
                out[(size_t)(rbase + e) * N + col] =
                    (float)acc[mi][ni][e] * xsv[e] * wsv + bv;
        }
    }
}

extern "C" void kernel_launch(void* const* d_in, const int* in_sizes, int n_in,
                              void* d_out, int out_size, void* d_ws, size_t ws_size,
                              hipStream_t stream) {
    const float* x      = (const float*)d_in[0];
    const float* weight = (const float*)d_in[1];
    const float* bias   = (const float*)d_in[2];
    float* out = (float*)d_out;

    const int N = in_sizes[2];
    const int K = in_sizes[1] / N;
    const int M = in_sizes[0] / K;

    signed char* xq = (signed char*)d_ws;
    signed char* wq = xq + (size_t)M * K;
    float* xs = (float*)(wq + (size_t)N * K);
    float* wsc = xs + M;

    quantize_rows_kernel<<<N, 256, 0, stream>>>(weight, wq, wsc, K);
    quantize_rows_kernel<<<M, 256, 0, stream>>>(x, xq, xs, K);

    dim3 grid(N / 128, M / 128);
    int8_gemm_wmma_kernel<<<grid, 128, 0, stream>>>(xq, wq, xs, wsc, bias, out, M, N, K);
}
